// Model_SAGE_2834678415939
// MI455X (gfx1250) — matrix operand order repaired and checked by reading, not compiled, not run on hardware
//
#include <hip/hip_runtime.h>
#include <hip/hip_bf16.h>
#include <math.h>

typedef __bf16 bf16;
typedef __attribute__((ext_vector_type(16))) __bf16 bf16x16;
typedef __attribute__((ext_vector_type(8)))  __bf16 bf16x8;
typedef __attribute__((ext_vector_type(8)))  float  f32x8;

#define WMMA_BF16(a,b,c) __builtin_amdgcn_wmma_f32_16x16x32_bf16(false,(a),false,(b),(short)0,(c),false,false)

// ---------------------------------------------------------------------------
// Model dims
// ---------------------------------------------------------------------------
#define BATCH 64
#define NNODE 512
#define LDIM  1024
#define NROWS (BATCH*NNODE)      // 32768
#define DFF   256
#define MODE_N 8
#define EDGES (BATCH*NNODE*8)    // 262144
#define FIN   514
#define FINP  544                // padded to mult of 32
#define DIMC  128

// ---------------------------------------------------------------------------
// WMMA GEMM: C(MxNp,f32) = act( A(MxKp,bf16 row-major) * Bt(NpxKp,bf16)^T + bias [+C] )
// flags: bit0 = accumulate into existing C ; act = (flags>>1): 0 none, 1 relu, 2 tanh
// M, Np multiples of 32; Kp multiple of 32. One wave computes a 32x32 tile.
// ---------------------------------------------------------------------------
__global__ __launch_bounds__(256) void k_wmma_gemm(
    const bf16* __restrict__ A, const bf16* __restrict__ Bt,
    float* __restrict__ C, const float* __restrict__ bias,
    int M, int Np, int Kp, int flags)
{
  const int nt    = Np >> 5;
  const int tiles = (M >> 5) * nt;
  const int wave  = blockIdx.x * 8 + (threadIdx.x >> 5);
  if (wave >= tiles) return;
  const int tm = wave / nt;
  const int tn = wave - tm * nt;
  const int lane = threadIdx.x & 31;
  const int l  = lane & 15;
  const int hi = lane >> 4;
  const int i0 = tm << 5;
  const int j0 = tn << 5;

  // A fragment (16x32 bf16): lane l(row M=l): elems[0..15]=K kk+hi*16+(0..15)
  // (contiguous, matching B)
  const bf16* pa0 = A  + (size_t)(i0 + l) * Kp + (hi << 4);
  const bf16* pa1 = pa0 + ((size_t)Kp << 4);
  // B fragment (32x16): lane l = column N=j0+l, K kk+hi*16 .. +16 contiguous in Bt(NxK)
  const bf16* pb0 = Bt + (size_t)(j0 + l) * Kp + (hi << 4);
  const bf16* pb1 = pb0 + ((size_t)Kp << 4);

  f32x8 c00 = {0.f,0.f,0.f,0.f,0.f,0.f,0.f,0.f};
  f32x8 c01 = c00, c10 = c00, c11 = c00;
  const int r0 = i0 + (hi << 3);
  if (flags & 1) {
#pragma unroll
    for (int r = 0; r < 8; ++r) {
      c00[r] = C[(size_t)(r0 + r) * Np + j0 + l];
      c01[r] = C[(size_t)(r0 + r) * Np + j0 + 16 + l];
      c10[r] = C[(size_t)(r0 + 16 + r) * Np + j0 + l];
      c11[r] = C[(size_t)(r0 + 16 + r) * Np + j0 + 16 + l];
    }
  }
  for (int kk = 0; kk < Kp; kk += 32) {
    bf16x8 al0 = *(const bf16x8*)(pa0 + kk);
    bf16x8 ah0 = *(const bf16x8*)(pa0 + kk + 8);
    bf16x8 al1 = *(const bf16x8*)(pa1 + kk);
    bf16x8 ah1 = *(const bf16x8*)(pa1 + kk + 8);
    bf16x16 a0 = __builtin_shufflevector(al0, ah0, 0,1,2,3,4,5,6,7,8,9,10,11,12,13,14,15);
    bf16x16 a1 = __builtin_shufflevector(al1, ah1, 0,1,2,3,4,5,6,7,8,9,10,11,12,13,14,15);
    bf16x16 b0 = *(const bf16x16*)(pb0 + kk);
    bf16x16 b1 = *(const bf16x16*)(pb1 + kk);
    c00 = WMMA_BF16(a0, b0, c00);
    c01 = WMMA_BF16(a0, b1, c01);
    c10 = WMMA_BF16(a1, b0, c10);
    c11 = WMMA_BF16(a1, b1, c11);
  }
  const int act = (flags >> 1) & 3;
  const float b0v = bias ? bias[j0 + l]      : 0.f;
  const float b1v = bias ? bias[j0 + 16 + l] : 0.f;
#pragma unroll
  for (int r = 0; r < 8; ++r) {
    float v00 = c00[r] + b0v, v01 = c01[r] + b1v;
    float v10 = c10[r] + b0v, v11 = c11[r] + b1v;
    if (act == 1) {
      v00 = fmaxf(v00, 0.f); v01 = fmaxf(v01, 0.f);
      v10 = fmaxf(v10, 0.f); v11 = fmaxf(v11, 0.f);
    } else if (act == 2) {
      v00 = tanhf(v00); v01 = tanhf(v01);
      v10 = tanhf(v10); v11 = tanhf(v11);
    }
    C[(size_t)(r0 + r) * Np + j0 + l]           = v00;
    C[(size_t)(r0 + r) * Np + j0 + 16 + l]      = v01;
    C[(size_t)(r0 + 16 + r) * Np + j0 + l]      = v10;
    C[(size_t)(r0 + 16 + r) * Np + j0 + 16 + l] = v11;
  }
}

// ---------------------------------------------------------------------------
// Helpers: conversions / weight prep
// ---------------------------------------------------------------------------
__global__ void k_cvt_bf16(const float* __restrict__ x, bf16* __restrict__ y, size_t n) {
  size_t i = (size_t)blockIdx.x * blockDim.x + threadIdx.x;
  if (i < n) y[i] = (bf16)x[i];
}

// W (KxN f32 row-major) -> Bt (NpxKp bf16, transposed, zero padded)
__global__ void k_prep_wt(const float* __restrict__ w, bf16* __restrict__ wt,
                          int K, int N, int Kp, int Np) {
  size_t i = (size_t)blockIdx.x * blockDim.x + threadIdx.x;
  if (i >= (size_t)Kp * Np) return;
  int n = (int)(i / Kp), k = (int)(i - (size_t)n * Kp);
  float v = (n < N && k < K) ? w[(size_t)k * N + n] : 0.f;
  wt[i] = (bf16)v;
}

__global__ void k_pad_vec(const float* __restrict__ b, float* __restrict__ o, int n, int np) {
  int i = blockIdx.x * blockDim.x + threadIdx.x;
  if (i < np) o[i] = (i < n) ? b[i] : 0.f;
}

__global__ void k_seed_bcast(const float* __restrict__ seeds, float* __restrict__ q0) {
  size_t i = (size_t)blockIdx.x * blockDim.x + threadIdx.x;
  if (i >= (size_t)BATCH * MODE_N * LDIM) return;
  int row = (int)(i >> 10), col = (int)(i & 1023);
  q0[i] = seeds[((row & (MODE_N - 1)) << 10) + col];
}

// ---------------------------------------------------------------------------
// x = LN(x + d) * g + be   (row-wise, C=1024)
// ---------------------------------------------------------------------------
__global__ __launch_bounds__(256) void k_add_ln(
    float* __restrict__ x, const float* __restrict__ d,
    const float* __restrict__ g, const float* __restrict__ be, int C)
{
  __shared__ float red[256], red2[256];
  const size_t ro = (size_t)blockIdx.x * C;
  float s = 0.f, s2 = 0.f;
  for (int c = threadIdx.x; c < C; c += 256) {
    float v = x[ro + c] + d[ro + c];
    s += v; s2 += v * v;
  }
  red[threadIdx.x] = s; red2[threadIdx.x] = s2;
  __syncthreads();
  for (int off = 128; off > 0; off >>= 1) {
    if ((int)threadIdx.x < off) {
      red[threadIdx.x]  += red[threadIdx.x + off];
      red2[threadIdx.x] += red2[threadIdx.x + off];
    }
    __syncthreads();
  }
  const float mean = red[0] / C;
  const float var  = red2[0] / C - mean * mean;
  const float rstd = rsqrtf(var + 1e-5f);
  for (int c = threadIdx.x; c < C; c += 256)
    x[ro + c] = (x[ro + c] + d[ro + c] - mean) * rstd * g[c] + be[c];
}

// ---------------------------------------------------------------------------
// MHA core (H=4, DH=4), qkv stored (rows,32) f32 with head h at cols h*4..h*4+3.
// One block per (batch,head); K/V staged in LDS; online softmax per query.
// ---------------------------------------------------------------------------
__global__ __launch_bounds__(256) void k_attn(
    const float* __restrict__ Q, const float* __restrict__ K,
    const float* __restrict__ V, float* __restrict__ O, int Lq, int Lk)
{
  __shared__ float kf[512 * 4];
  __shared__ float vf[512 * 4];
  const int b = blockIdx.x >> 2;
  const int h = blockIdx.x & 3;
  for (int j = threadIdx.x; j < Lk; j += 256) {
    *(float4*)&kf[j * 4] = *(const float4*)(K + ((size_t)b * Lk + j) * 32 + h * 4);
    *(float4*)&vf[j * 4] = *(const float4*)(V + ((size_t)b * Lk + j) * 32 + h * 4);
  }
  __syncthreads();
  for (int q = threadIdx.x; q < Lq; q += 256) {
    const float* qp = Q + ((size_t)b * Lq + q) * 32 + h * 4;
    float q0 = qp[0], q1 = qp[1], q2 = qp[2], q3 = qp[3];
    float mr = -1e30f, ls = 0.f, a0 = 0.f, a1 = 0.f, a2 = 0.f, a3 = 0.f;
    for (int j = 0; j < Lk; ++j) {
      float s = 0.5f * (q0*kf[j*4] + q1*kf[j*4+1] + q2*kf[j*4+2] + q3*kf[j*4+3]);
      float nm = fmaxf(mr, s);
      float sc = expf(mr - nm);
      float w  = expf(s - nm);
      ls = ls * sc + w;
      a0 = a0 * sc + w * vf[j*4];   a1 = a1 * sc + w * vf[j*4+1];
      a2 = a2 * sc + w * vf[j*4+2]; a3 = a3 * sc + w * vf[j*4+3];
      mr = nm;
    }
    float inv = 1.f / ls;
    float* op = O + ((size_t)b * Lq + q) * 32 + h * 4;
    op[0] = a0 * inv; op[1] = a1 * inv; op[2] = a2 * inv; op[3] = a3 * inv;
  }
}

// ---------------------------------------------------------------------------
// 512-point radix-2 FFT per row -> feat = [ |rfft| (257), angle (257), 0-pad to 544 ]
// ---------------------------------------------------------------------------
__global__ __launch_bounds__(128) void k_fft_feat(const float* __restrict__ acc,
                                                  float* __restrict__ feat)
{
  __shared__ float re[512], im[512];
  const float* xr = acc + (size_t)blockIdx.x * 1024;
  for (int i = threadIdx.x; i < 512; i += 128) {
    unsigned r = __brev((unsigned)i) >> 23;   // 9-bit bit reversal
    re[r] = xr[i];
    im[r] = 0.f;
  }
  __syncthreads();
  for (int s = 1; s <= 9; ++s) {
    const int m = 1 << s, half = m >> 1;
    for (int t = threadIdx.x; t < 256; t += 128) {
      int grp = t / half, pos = t - grp * half;
      int i1 = grp * m + pos, i2 = i1 + half;
      float ang = -6.283185307179586f * (float)pos / (float)m;
      float wr, wi;
      __sincosf(ang, &wi, &wr);
      float tr = wr * re[i2] - wi * im[i2];
      float ti = wr * im[i2] + wi * re[i2];
      float ur = re[i1], ui = im[i1];
      re[i1] = ur + tr; im[i1] = ui + ti;
      re[i2] = ur - tr; im[i2] = ui - ti;
    }
    __syncthreads();
  }
  float* fr = feat + (size_t)blockIdx.x * FINP;
  for (int k = threadIdx.x; k < 257; k += 128) {
    float a = re[k], b = im[k];
    fr[k]       = sqrtf(a * a + b * b);
    fr[257 + k] = atan2f(b, a);
  }
  for (int k = FIN + threadIdx.x; k < FINP; k += 128) fr[k] = 0.f;
}

// ---------------------------------------------------------------------------
// segment max: hn[dst[e]] = max(hn[dst[e]], m[src[e]])  (values >= 0 -> uint cmp)
// ---------------------------------------------------------------------------
__global__ void k_seg_max(const float* __restrict__ m, const int* __restrict__ src,
                          const int* __restrict__ dst, unsigned* __restrict__ hn, int C)
{
  const int e = blockIdx.x;
  const size_t so = (size_t)src[e] * C;
  const size_t dofs = (size_t)dst[e] * C;
  for (int c = threadIdx.x; c < C; c += 256)
    atomicMax(hn + dofs + c, __float_as_uint(m[so + c]));
}

__global__ void k_colmax(const float* __restrict__ phi, unsigned* __restrict__ cm, int nrows) {
  size_t i = (size_t)blockIdx.x * blockDim.x + threadIdx.x;
  if (i >= (size_t)nrows * MODE_N) return;
  int r = (int)(i >> 3), c = (int)(i & 7);
  atomicMax(&cm[c], __float_as_uint(fabsf(phi[(size_t)r * 32 + c])));
}

__global__ void k_phidiv(const float* __restrict__ phi, const unsigned* __restrict__ cm,
                         float* __restrict__ out, int nrows) {
  size_t i = (size_t)blockIdx.x * blockDim.x + threadIdx.x;
  if (i >= (size_t)nrows * MODE_N) return;
  int r = (int)(i >> 3), c = (int)(i & 7);
  out[(size_t)r * MODE_N + c] = phi[(size_t)r * 32 + c] / __uint_as_float(cm[c]);
}

// ---------------------------------------------------------------------------
// Host-side orchestration
// ---------------------------------------------------------------------------
static void gemm(hipStream_t s, const bf16* A, const bf16* Bt, float* C,
                 const float* bias, int M, int Np, int Kp, int flags) {
  int tiles = (M >> 5) * (Np >> 5);
  k_wmma_gemm<<<(tiles + 7) / 8, 256, 0, s>>>(A, Bt, C, bias, M, Np, Kp, flags);
}
static void cvt(hipStream_t s, const float* x, bf16* y, size_t n) {
  k_cvt_bf16<<<(unsigned)((n + 255) / 256), 256, 0, s>>>(x, y, n);
}
static void prepw(hipStream_t s, const float* w, bf16* wt, int K, int N, int Kp, int Np) {
  size_t n = (size_t)Kp * Np;
  k_prep_wt<<<(unsigned)((n + 255) / 256), 256, 0, s>>>(w, wt, K, N, Kp, Np);
}

// Input flattening: top-level dict insertion order (acc_Y, params, src, dst);
// params flattened as JAX pytree (dict keys sorted recursively, lists in order).
enum {
  IN_ACC = 0,
  IN_C1 = 1,   // Wn, Wp, Ws, b, bp
  IN_C2 = 6,
  IN_C3 = 11,
  IN_DEC0 = 16,  // W1,W2,Wk,Wo,Wq,Wv,b1,b2,be1,be2,g1,g2
  IN_DEC1 = 28,
  IN_ENC0 = 40,
  IN_ENC1 = 52,
  IN_MLP = 64,   // W1,W2,W3,b1,b2,b3
  IN_PMA_W1 = 70, IN_PMA_W2 = 71, IN_PMA_B1 = 72, IN_PMA_B2 = 73,
  IN_PMA_MHA = 74,
  IN_PMA_SEEDS = 86,
  IN_SRC = 87, IN_DST = 88
};
enum { MW1 = 0, MW2, MWK, MWO, MWQ, MWV, MB1, MB2, MBE1, MBE2, MG1, MG2 };

struct MabDev {
  bf16 *WqT, *WkT, *WvT, *WoT, *W1T, *W2T;
  const float *b1, *b2, *g1, *be1, *g2, *be2;
};

struct Scratch {
  float *Q32, *K32, *V32, *O32, *TMP, *FF1;
  bf16  *OBF, *FF1BF;
};

static void run_mab(hipStream_t s, float* X, bf16* XBF, const float* MEM, bf16* MEMBF,
                    int rows, int Lq, int mrows, int Lk, const MabDev& W, const Scratch& sc)
{
  cvt(s, X, XBF, (size_t)rows * LDIM);
  if (MEM != X) cvt(s, MEM, MEMBF, (size_t)mrows * LDIM);
  gemm(s, XBF,   W.WqT, sc.Q32, nullptr, rows,  32, LDIM, 0);
  gemm(s, MEMBF, W.WkT, sc.K32, nullptr, mrows, 32, LDIM, 0);
  gemm(s, MEMBF, W.WvT, sc.V32, nullptr, mrows, 32, LDIM, 0);
  k_attn<<<(rows / Lq) * 4, 256, 0, s>>>(sc.Q32, sc.K32, sc.V32, sc.O32, Lq, Lk);
  cvt(s, sc.O32, sc.OBF, (size_t)rows * 32);
  gemm(s, sc.OBF, W.WoT, sc.TMP, nullptr, rows, LDIM, 32, 0);
  k_add_ln<<<rows, 256, 0, s>>>(X, sc.TMP, W.g1, W.be1, LDIM);
  cvt(s, X, XBF, (size_t)rows * LDIM);
  gemm(s, XBF, W.W1T, sc.FF1, W.b1, rows, DFF, LDIM, 2 /*relu*/);
  cvt(s, sc.FF1, sc.FF1BF, (size_t)rows * DFF);
  gemm(s, sc.FF1BF, W.W2T, sc.TMP, W.b2, rows, LDIM, DFF, 0);
  k_add_ln<<<rows, 256, 0, s>>>(X, sc.TMP, W.g2, W.be2, LDIM);
}

extern "C" void kernel_launch(void* const* d_in, const int* in_sizes, int n_in,
                              void* d_out, int out_size, void* d_ws, size_t ws_size,
                              hipStream_t stream)
{
  (void)in_sizes; (void)n_in; (void)out_size; (void)ws_size;
  const float* acc = (const float*)d_in[IN_ACC];
  const int* src = (const int*)d_in[IN_SRC];
  const int* dst = (const int*)d_in[IN_DST];
  float* out = (float*)d_out;

  // ---------------- workspace layout (bump allocator, 256B aligned) --------
  char* base = (char*)d_ws;
  size_t off = 0;
  auto take = [&](size_t bytes) -> char* {
    char* r = base + off;
    off = (off + bytes + 255) & ~(size_t)255;
    return r;
  };
  float* X     = (float*)take((size_t)NROWS * LDIM * 4);   // 128MB; reused for FEAT
  float* FEAT  = X;                                        // NROWS x 544 fits
  bf16*  XBF   = (bf16*) take((size_t)NROWS * LDIM * 2);   // 64MB
  float* TMP   = (float*)take((size_t)NROWS * LDIM * 4);   // 128MB (attn-out / mem / m)
  float* Q32   = (float*)take((size_t)NROWS * 32 * 4);
  float* K32   = (float*)take((size_t)NROWS * 32 * 4);
  float* V32   = (float*)take((size_t)NROWS * 32 * 4);
  float* O32   = (float*)take((size_t)NROWS * 32 * 4);
  bf16*  OBF   = (bf16*) take((size_t)NROWS * 32 * 2);
  float* FF1   = (float*)take((size_t)NROWS * DFF * 4);    // 32MB
  bf16*  FF1BF = (bf16*) take((size_t)NROWS * FINP * 2);   // sized for FEATBF reuse
  bf16*  FEATBF = FF1BF;
  float* HN    = (float*)take((size_t)NROWS * FINP * 4);   // 68MB
  bf16*  HNBF  = (bf16*) take((size_t)NROWS * FINP * 2);
  float* H1    = (float*)take((size_t)NROWS * DIMC * 4);
  float* H2    = (float*)take((size_t)NROWS * DIMC * 4);
  bf16*  HBF   = (bf16*) take((size_t)NROWS * DIMC * 2);
  float* PHI   = (float*)take((size_t)NROWS * 32 * 4);
  float* Q0    = (float*)take((size_t)BATCH * MODE_N * LDIM * 4);
  bf16*  Q0BF  = (bf16*) take((size_t)BATCH * MODE_N * LDIM * 2);
  unsigned* CM = (unsigned*)take(256);
  float* BP1P  = (float*)take(FINP * 4);
  float* B3P   = (float*)take(32 * 4);

  // ---------------- weight prep (f32 -> transposed, padded bf16) -----------
  auto prep_mab = [&](int b) -> MabDev {
    auto W = [&](int k) { return (const float*)d_in[b + k]; };
    MabDev m;
    m.WqT = (bf16*)take(32 * 1024 * 2);   prepw(stream, W(MWQ), m.WqT, 1024, 16, 1024, 32);
    m.WkT = (bf16*)take(32 * 1024 * 2);   prepw(stream, W(MWK), m.WkT, 1024, 16, 1024, 32);
    m.WvT = (bf16*)take(32 * 1024 * 2);   prepw(stream, W(MWV), m.WvT, 1024, 16, 1024, 32);
    m.WoT = (bf16*)take(1024 * 32 * 2);   prepw(stream, W(MWO), m.WoT, 16, 1024, 32, 1024);
    m.W1T = (bf16*)take(256 * 1024 * 2);  prepw(stream, W(MW1), m.W1T, 1024, 256, 1024, 256);
    m.W2T = (bf16*)take(1024 * 256 * 2);  prepw(stream, W(MW2), m.W2T, 256, 1024, 256, 1024);
    m.b1 = W(MB1); m.b2 = W(MB2);
    m.g1 = W(MG1); m.be1 = W(MBE1); m.g2 = W(MG2); m.be2 = W(MBE2);
    return m;
  };
  MabDev enc0 = prep_mab(IN_ENC0), enc1 = prep_mab(IN_ENC1);
  MabDev pmaM = prep_mab(IN_PMA_MHA);
  MabDev dec0 = prep_mab(IN_DEC0), dec1 = prep_mab(IN_DEC1);

  bf16* pW1T = (bf16*)take(256 * 1024 * 2);
  prepw(stream, (const float*)d_in[IN_PMA_W1], pW1T, 1024, 256, 1024, 256);
  bf16* pW2T = (bf16*)take(1024 * 256 * 2);
  prepw(stream, (const float*)d_in[IN_PMA_W2], pW2T, 256, 1024, 256, 1024);

  bf16* Wp1T = (bf16*)take((size_t)FINP * FINP * 2);
  prepw(stream, (const float*)d_in[IN_C1 + 1], Wp1T, FIN, FIN, FINP, FINP);
  bf16* Ws1T = (bf16*)take((size_t)DIMC * FINP * 2);
  prepw(stream, (const float*)d_in[IN_C1 + 2], Ws1T, FIN, DIMC, FINP, DIMC);
  bf16* Wn1T = (bf16*)take((size_t)DIMC * FINP * 2);
  prepw(stream, (const float*)d_in[IN_C1 + 0], Wn1T, FIN, DIMC, FINP, DIMC);

  auto prep_sq = [&](int idx) {  // DIMCxDIMC weight -> transposed bf16
    bf16* t = (bf16*)take((size_t)DIMC * DIMC * 2);
    prepw(stream, (const float*)d_in[idx], t, DIMC, DIMC, DIMC, DIMC);
    return t;
  };
  bf16 *Wp2T = prep_sq(IN_C2 + 1), *Ws2T = prep_sq(IN_C2 + 2), *Wn2T = prep_sq(IN_C2 + 0);
  bf16 *Wp3T = prep_sq(IN_C3 + 1), *Ws3T = prep_sq(IN_C3 + 2), *Wn3T = prep_sq(IN_C3 + 0);
  bf16 *mW1T = prep_sq(IN_MLP + 0), *mW2T = prep_sq(IN_MLP + 1);
  bf16* mW3T = (bf16*)take(32 * DIMC * 2);
  prepw(stream, (const float*)d_in[IN_MLP + 2], mW3T, DIMC, MODE_N, DIMC, 32);

  k_pad_vec<<<3, 256, 0, stream>>>((const float*)d_in[IN_C1 + 4], BP1P, FIN, FINP);
  k_pad_vec<<<1, 32, 0, stream>>>((const float*)d_in[IN_MLP + 5], B3P, MODE_N, 32);

  Scratch sc{Q32, K32, V32, O32, TMP, FF1, OBF, FF1BF};

  // ======================= SetTransformer =================================
  hipMemcpyAsync(X, acc, (size_t)NROWS * LDIM * 4, hipMemcpyDeviceToDevice, stream);
  run_mab(stream, X, XBF, X, XBF, NROWS, NNODE, NROWS, NNODE, enc0, sc);
  run_mab(stream, X, XBF, X, XBF, NROWS, NNODE, NROWS, NNODE, enc1, sc);

  // PMA: mem = rFF(x)
  cvt(stream, X, XBF, (size_t)NROWS * LDIM);
  gemm(stream, XBF, pW1T, FF1, (const float*)d_in[IN_PMA_B1], NROWS, DFF, LDIM, 2);
  cvt(stream, FF1, FF1BF, (size_t)NROWS * DFF);
  gemm(stream, FF1BF, pW2T, TMP, (const float*)d_in[IN_PMA_B2], NROWS, LDIM, DFF, 0);

  k_seed_bcast<<<(BATCH * MODE_N * LDIM + 255) / 256, 256, 0, stream>>>(
      (const float*)d_in[IN_PMA_SEEDS], Q0);
  run_mab(stream, Q0, Q0BF, TMP, XBF, BATCH * MODE_N, MODE_N, NROWS, NNODE, pmaM, sc);
  run_mab(stream, Q0, Q0BF, Q0, Q0BF, BATCH * MODE_N, MODE_N, BATCH * MODE_N, MODE_N, dec0, sc);
  run_mab(stream, Q0, Q0BF, Q0, Q0BF, BATCH * MODE_N, MODE_N, BATCH * MODE_N, MODE_N, dec1, sc);
  // q output: reshape(B,8192)->(B,1024,8) preserves memory order -> direct copy
  hipMemcpyAsync(out, Q0, (size_t)BATCH * MODE_N * LDIM * 4, hipMemcpyDeviceToDevice, stream);

  // ======================= FFT features ===================================
  k_fft_feat<<<NROWS, 128, 0, stream>>>(acc, FEAT);

  // ======================= SAGE conv1 (tanh) ==============================
  cvt(stream, FEAT, FEATBF, (size_t)NROWS * FINP);
  gemm(stream, FEATBF, Wp1T, TMP, BP1P, NROWS, FINP, FINP, 2 /*relu*/);
  hipMemsetAsync(HN, 0, (size_t)NROWS * FINP * 4, stream);
  k_seg_max<<<EDGES, 256, 0, stream>>>(TMP, src, dst, (unsigned*)HN, FINP);
  gemm(stream, FEATBF, Ws1T, H1, (const float*)d_in[IN_C1 + 3], NROWS, DIMC, FINP, 0);
  cvt(stream, HN, HNBF, (size_t)NROWS * FINP);
  gemm(stream, HNBF, Wn1T, H1, nullptr, NROWS, DIMC, FINP, 1 | (2 << 1) /*acc+tanh*/);

  // ======================= SAGE conv2 (tanh) ==============================
  cvt(stream, H1, HBF, (size_t)NROWS * DIMC);
  gemm(stream, HBF, Wp2T, TMP, (const float*)d_in[IN_C2 + 4], NROWS, DIMC, DIMC, 2);
  hipMemsetAsync(HN, 0, (size_t)NROWS * DIMC * 4, stream);
  k_seg_max<<<EDGES, 256, 0, stream>>>(TMP, src, dst, (unsigned*)HN, DIMC);
  gemm(stream, HBF, Ws2T, H2, (const float*)d_in[IN_C2 + 3], NROWS, DIMC, DIMC, 0);
  cvt(stream, HN, HNBF, (size_t)NROWS * DIMC);
  gemm(stream, HNBF, Wn2T, H2, nullptr, NROWS, DIMC, DIMC, 1 | (2 << 1));

  // ======================= SAGE conv3 (linear) ============================
  cvt(stream, H2, HBF, (size_t)NROWS * DIMC);
  gemm(stream, HBF, Wp3T, TMP, (const float*)d_in[IN_C3 + 4], NROWS, DIMC, DIMC, 2);
  hipMemsetAsync(HN, 0, (size_t)NROWS * DIMC * 4, stream);
  k_seg_max<<<EDGES, 256, 0, stream>>>(TMP, src, dst, (unsigned*)HN, DIMC);
  gemm(stream, HBF, Ws3T, H1, (const float*)d_in[IN_C3 + 3], NROWS, DIMC, DIMC, 0);
  cvt(stream, HN, HNBF, (size_t)NROWS * DIMC);
  gemm(stream, HNBF, Wn3T, H1, nullptr, NROWS, DIMC, DIMC, 1 /*acc only*/);

  // ======================= MLP + normalize ================================
  cvt(stream, H1, HBF, (size_t)NROWS * DIMC);
  gemm(stream, HBF, mW1T, H2, (const float*)d_in[IN_MLP + 3], NROWS, DIMC, DIMC, 4 /*tanh*/);
  cvt(stream, H2, HBF, (size_t)NROWS * DIMC);
  gemm(stream, HBF, mW2T, H1, (const float*)d_in[IN_MLP + 4], NROWS, DIMC, DIMC, 4);
  cvt(stream, H1, HBF, (size_t)NROWS * DIMC);
  gemm(stream, HBF, mW3T, PHI, B3P, NROWS, 32, DIMC, 0);

  hipMemsetAsync(CM, 0, 32 * 4, stream);
  k_colmax<<<(NROWS * MODE_N + 255) / 256, 256, 0, stream>>>(PHI, CM, NROWS);
  k_phidiv<<<(NROWS * MODE_N + 255) / 256, 256, 0, stream>>>(
      PHI, CM, out + (size_t)BATCH * MODE_N * LDIM, NROWS);
}
